// HebbianConv2d_49615462203908
// MI455X (gfx1250) — matrix operand order repaired and checked by reading, not compiled, not run on hardware
//
#include <hip/hip_runtime.h>

// ---------------- problem constants (from reference) ----------------
#define B_SZ   32
#define C_SZ   64
#define H_SZ   64
#define W_SZ   64
#define O_CH   128
#define KTOT   576          // C*3*3
#define HO     62
#define WO     62
#define L_OUT  3844         // HO*WO
#define BL     123008       // B_SZ*L_OUT  (= 3844 * 32)
#define NUNITS 1922         // BL / 64 : K-step pairs for dw GEMM
#define KSPLIT 16
#define KCHUNKU 121         // ceil(NUNITS/KSPLIT)
#define MTILES 121          // ceil(L_OUT/32)
#define Y_ELEMS (B_SZ * O_CH * L_OUT)
#define DW_ELEMS (O_CH * KTOT)
#define XPLANE (C_SZ * H_SZ * W_SZ)   // 262144

typedef __bf16 v16bf __attribute__((ext_vector_type(16)));
typedef __bf16 v8bf  __attribute__((ext_vector_type(8)));
typedef __bf16 v4bf  __attribute__((ext_vector_type(4)));
typedef float  v8f   __attribute__((ext_vector_type(8)));

// ---------------- workspace layout (bytes, 256-aligned) ----------------
#define WS_XBF    0
#define WS_BPACK  16777216
#define WS_NORMS  (WS_BPACK + 147456)
#define WS_RSUM   (WS_NORMS + 512)
#define WS_R2SUM  (WS_RSUM + 512)
#define WS_CRBF   (WS_R2SUM + 512)
#define WS_DWACC  (WS_CRBF + 31490048)

// ---------------- kernel 0: zero accumulators ----------------
__global__ void hc_init_kernel(float* dwacc, float* rsum, float* r2sum) {
    int i = blockIdx.x * 256 + threadIdx.x;
    if (i < DW_ELEMS) dwacc[i] = 0.0f;
    if (i < O_CH) { rsum[i] = 0.0f; r2sum[i] = 0.0f; }
}

// ---------------- kernel 1: per-filter L2 norms ----------------
__global__ __launch_bounds__(256) void hc_norm_kernel(const float* w, float* norms) {
    int o = blockIdx.x;
    __shared__ float red[256];
    float s = 0.0f;
    for (int k = threadIdx.x; k < KTOT; k += 256) {
        float v = w[o * KTOT + k];
        s += v * v;
    }
    red[threadIdx.x] = s;
    __syncthreads();
    for (int st = 128; st > 0; st >>= 1) {
        if (threadIdx.x < st) red[threadIdx.x] += red[threadIdx.x + st];
        __syncthreads();
    }
    if (threadIdx.x == 0) {
        float n = sqrtf(red[0]);
        norms[o] = (n == 0.0f) ? 1.0f : n;
    }
}

// ---------------- kernel 2: pack normalized weights into B-fragment layout ----
// bpack[((s*8 + nt)*32 + lane)*16 + h] = wn[o = nt*16+(lane&15)][K = s*32+(lane>>4)*8+(h<8?h:h+8)]
__global__ void hc_packw_kernel(const float* w, const float* norms, __bf16* bpack) {
    int idx = blockIdx.x * 256 + threadIdx.x;
    if (idx >= DW_ELEMS) return;
    int h    = idx & 15;
    int lane = (idx >> 4) & 31;
    int nt   = (idx >> 9) & 7;
    int s    = idx >> 12;
    int K = s * 32 + ((lane >> 4) * 8) + ((h < 8) ? h : (h + 8));
    int o = nt * 16 + (lane & 15);
    bpack[idx] = (__bf16)(w[o * KTOT + K] / norms[o]);
}

// ---------------- kernel 3: x fp32 -> bf16 ----------------
__global__ void hc_xconv_kernel(const float* x, __bf16* xbf, int n) {
    int i = blockIdx.x * 256 + threadIdx.x;
    if (i < n) xbf[i] = (__bf16)x[i];
}

// ---------------- kernel 4: conv as implicit GEMM (y = Xunf @ WnT + bias) ----
// block: (b, ltile of 32 positions); 8 waves; wave w -> channels [w*16, w*16+16)
// and BOTH 16-position M-tiles (B fragment reused across the two WMMAs).
__global__ __launch_bounds__(256) void hc_conv_gemm_kernel(
        const __bf16* __restrict__ xbf, const __bf16* __restrict__ bpack,
        const float* __restrict__ bias, float* __restrict__ y) {
    int b     = blockIdx.x;
    int lbase = blockIdx.y * 32;
    int tid   = threadIdx.x;
    int lane  = tid & 31;
    int wave  = tid >> 5;

    __shared__ __bf16 A_lds[2][32][16];

    v8f acc0, acc1;
    #pragma unroll
    for (int i = 0; i < 8; ++i) { acc0[i] = 0.0f; acc1[i] = 0.0f; }

    // staging: thread t fills A_lds[mt][s_lane][h0 .. h0+3] (4 contiguous halves,
    // which are 4 contiguous K values by construction of the A-fragment layout)
    int mt     = tid >> 7;                 // which M-tile
    int s_lane = (tid >> 2) & 31;
    int h0     = (tid & 3) * 4;            // 0,4,8,12
    int s_row  = s_lane & 15;
    int s_kbase = (s_lane >> 4) * 8;
    int kb0 = s_kbase + ((h0 < 8) ? h0 : (h0 + 8));   // K of half h0 at s=0

    int l0 = lbase + mt * 16 + s_row;
    if (l0 >= L_OUT) l0 = L_OUT - 1;       // clamp tail rows (stores masked later)
    int ho = l0 / WO;
    int wo = l0 - ho * WO;
    const __bf16* xb = xbf + (size_t)b * XPLANE + (size_t)ho * W_SZ + wo;

    // incremental K -> (c, rem) state, K = s*32 + kb0, rem = K % 9, c = K / 9
    int c   = kb0 / 9;
    int rem = kb0 - c * 9;

    for (int s = 0; s < 18; ++s) {
        __syncthreads();
        v4bf vals;
        int cu = c, ru = rem;
        #pragma unroll
        for (int u = 0; u < 4; ++u) {
            int ki = (ru * 11) >> 5;       // ru/3 for ru in [0,8]
            int kj = ru - ki * 3;
            vals[u] = xb[cu * (H_SZ * W_SZ) + ki * W_SZ + kj];
            ru += 1;
            if (ru >= 9) { ru = 0; cu += 1; }
        }
        *reinterpret_cast<v4bf*>(&A_lds[mt][s_lane][h0]) = vals;
        // advance K by 32: c += 3, rem += 5 with carry
        c += 3; rem += 5;
        if (rem >= 9) { rem -= 9; c += 1; }
        __syncthreads();

        v16bf a0 = *reinterpret_cast<const v16bf*>(&A_lds[0][lane][0]);
        v16bf a1 = *reinterpret_cast<const v16bf*>(&A_lds[1][lane][0]);
        v16bf bm = *reinterpret_cast<const v16bf*>(
                       bpack + ((size_t)(s * 8 + wave) * 32 + lane) * 16);
        acc0 = __builtin_amdgcn_wmma_f32_16x16x32_bf16(false, a0, false, bm, (short)0, acc0, false, false);
        acc1 = __builtin_amdgcn_wmma_f32_16x16x32_bf16(false, a1, false, bm, (short)0, acc1, false, false);
    }

    // D layout: vgpr r, lanes 0-15 -> M=r ; lanes 16-31 -> M=r+8 ; N = lane&15
    int o = wave * 16 + (lane & 15);
    float bv = bias[o];
    int moff = (lane >> 4) * 8;
    float* yb = y + ((size_t)b * O_CH + o) * L_OUT;
    #pragma unroll
    for (int r = 0; r < 8; ++r) {
        int l = lbase + moff + r;
        if (l < L_OUT) yb[l] = acc0[r] + bv;
        if (l + 16 < L_OUT) yb[l + 16] = acc1[r] + bv;
    }
}

// ---------------- kernel 5: softmax stats (r_sum[o], sum r^2[o]) ----------------
__global__ __launch_bounds__(256) void hc_softmax_stats_kernel(
        const float* __restrict__ y, float* rsum, float* r2sum) {
    __shared__ float lrs[O_CH];
    __shared__ float lr2[O_CH];
    int tid = threadIdx.x;
    if (tid < O_CH) { lrs[tid] = 0.0f; lr2[tid] = 0.0f; }
    __syncthreads();
    int p = blockIdx.x * 256 + tid;
    if (p < BL) {
        int b = p / L_OUT;
        int l = p - b * L_OUT;
        const float* col = y + (size_t)b * O_CH * L_OUT + l;
        float mx = -1e30f;
        for (int o = 0; o < O_CH; ++o) mx = fmaxf(mx, col[(size_t)o * L_OUT]);
        float se = 0.0f;
        for (int o = 0; o < O_CH; ++o) se += __expf(col[(size_t)o * L_OUT] - mx);
        float inv = 1.0f / se;
        for (int o = 0; o < O_CH; ++o) {
            float r = __expf(col[(size_t)o * L_OUT] - mx) * inv;
            atomicAdd(&lrs[o], r);        // ds_add_f32
            atomicAdd(&lr2[o], r * r);
        }
    }
    __syncthreads();
    if (tid < O_CH) {
        unsafeAtomicAdd(&rsum[tid], lrs[tid]);
        unsafeAtomicAdd(&r2sum[tid], lr2[tid]);
    }
}

// ---------------- kernel 6: build cr (bf16, [O][BL] row-major) ----------------
__global__ __launch_bounds__(256) void hc_cr_build_kernel(
        const float* __restrict__ y, const float* __restrict__ rsum,
        __bf16* __restrict__ crbf) {
    int p = blockIdx.x * 256 + threadIdx.x;
    if (p >= BL) return;
    int b = p / L_OUT;
    int l = p - b * L_OUT;
    const float* col = y + (size_t)b * O_CH * L_OUT + l;
    float mx = -1e30f;
    for (int o = 0; o < O_CH; ++o) mx = fmaxf(mx, col[(size_t)o * L_OUT]);
    float se = 0.0f;
    for (int o = 0; o < O_CH; ++o) se += __expf(col[(size_t)o * L_OUT] - mx);
    float inv = 1.0f / se;
    for (int o = 0; o < O_CH; ++o) {
        float r  = __expf(col[(size_t)o * L_OUT] - mx) * inv;
        float rs = rsum[o];
        rs = (rs == 0.0f) ? 1.0f : rs;
        crbf[(size_t)o * BL + p] = (__bf16)(r * r / rs);
    }
}

// ---------------- kernel 7: delta_w main GEMM  cr[O,BL] @ Xunf[BL,576] -------
// grid: (36 ntiles, KSPLIT); 8 waves; wave w -> O rows [w*16, w*16+16).
// K iterated in PAIRS of 32-steps (units); 2 WMMAs per barrier pair.
__global__ __launch_bounds__(256) void hc_dw_gemm_kernel(
        const __bf16* __restrict__ crbf, const __bf16* __restrict__ xbf,
        float* __restrict__ dwacc) {
    int nt = blockIdx.x;                   // 0..35
    int kc = blockIdx.y;                   // 0..KSPLIT-1
    int u0 = kc * KCHUNKU;
    int u1 = u0 + KCHUNKU;
    if (u1 > NUNITS) u1 = NUNITS;
    int tid  = threadIdx.x;
    int lane = tid & 31;
    int wave = tid >> 5;

    __shared__ __bf16 B_lds[2][32][16];

    v8f acc;
    #pragma unroll
    for (int i = 0; i < 8; ++i) acc[i] = 0.0f;

    // B staging: thread t fills B_lds[st][s_lane][h0..h0+3]
    // (B frag: lane col = lane&15, K = (lane>>4)*8 + (h<8 ? h : h+8) -> 4 contiguous K values)
    int st     = tid >> 7;                 // which step of the pair
    int s_lane = (tid >> 2) & 31;
    int h0     = (tid & 3) * 4;
    int ncol   = nt * 16 + (s_lane & 15);
    int cc   = ncol / 9;
    int rem  = ncol - cc * 9;
    int ki   = rem / 3;
    int kj   = rem - ki * 3;
    const __bf16* xcol = xbf + ((size_t)cc * H_SZ + ki) * W_SZ + kj;

    // incremental position state for p = (2u+st)*32 + (s_lane>>4)*8 + (h0<8 ? h0 : h0+8)
    int p0 = (2 * u0 + st) * 32 + ((s_lane >> 4) * 8) + ((h0 < 8) ? h0 : (h0 + 8));
    int bb = p0 / L_OUT;
    int lr = p0 - bb * L_OUT;
    int ho = lr / WO;
    int wo = lr - ho * WO;

    // A pointers (A frag: lane row = lane&15, kbase = (lane>>4)*8)
    int o = wave * 16 + (lane & 15);
    const __bf16* arow = crbf + (size_t)o * BL;
    int a_kb = (lane >> 4) * 8;

    for (int u = u0; u < u1; ++u) {
        __syncthreads();
        v4bf vals;
        #pragma unroll
        for (int q = 0; q < 4; ++q) {
            int wou = wo + q, hou = ho, bbu = bb;
            if (wou >= WO) { wou -= WO; hou += 1; }
            if (hou >= HO) { hou -= HO; bbu += 1; }
            vals[q] = xcol[(size_t)bbu * XPLANE + hou * W_SZ + wou];
        }
        *reinterpret_cast<v4bf*>(&B_lds[st][s_lane][h0]) = vals;
        // advance p by 64: wo += 64 with up to two carries into ho, one into bb
        wo += 64;
        if (wo >= WO) { wo -= WO; ho += 1; }
        if (wo >= WO) { wo -= WO; ho += 1; }
        if (ho >= HO) { ho -= HO; bb += 1; }
        __syncthreads();

        int k0 = u * 64 + a_kb;
        v8bf alo0 = *reinterpret_cast<const v8bf*>(arow + k0);
        v8bf ahi0 = *reinterpret_cast<const v8bf*>(arow + k0 + 16);
        v8bf alo1 = *reinterpret_cast<const v8bf*>(arow + k0 + 32);
        v8bf ahi1 = *reinterpret_cast<const v8bf*>(arow + k0 + 48);
        v16bf a0 = __builtin_shufflevector(alo0, ahi0,
                     0, 1, 2, 3, 4, 5, 6, 7, 8, 9, 10, 11, 12, 13, 14, 15);
        v16bf a1 = __builtin_shufflevector(alo1, ahi1,
                     0, 1, 2, 3, 4, 5, 6, 7, 8, 9, 10, 11, 12, 13, 14, 15);
        v16bf bm0 = *reinterpret_cast<const v16bf*>(&B_lds[0][lane][0]);
        v16bf bm1 = *reinterpret_cast<const v16bf*>(&B_lds[1][lane][0]);
        acc = __builtin_amdgcn_wmma_f32_16x16x32_bf16(false, a0, false, bm0, (short)0, acc, false, false);
        acc = __builtin_amdgcn_wmma_f32_16x16x32_bf16(false, a1, false, bm1, (short)0, acc, false, false);
    }

    int col  = nt * 16 + (lane & 15);
    int moff = (lane >> 4) * 8;
    #pragma unroll
    for (int r = 0; r < 8; ++r) {
        int oo = wave * 16 + moff + r;
        unsafeAtomicAdd(&dwacc[oo * KTOT + col], acc[r]);
    }
}

// ---------------- kernel 8: finalize delta_w = dwacc - (sum r^2 / r_sum) * w ---
__global__ void hc_finalize_kernel(const float* dwacc, const float* rsum,
                                   const float* r2sum, const float* w,
                                   float* out) {
    int idx = blockIdx.x * 256 + threadIdx.x;
    if (idx >= DW_ELEMS) return;
    int o = idx / KTOT;
    float rs = rsum[o];
    rs = (rs == 0.0f) ? 1.0f : rs;
    float crsum = r2sum[o] / rs;
    out[idx] = dwacc[idx] - crsum * w[idx];
}

// ---------------- launcher ----------------
extern "C" void kernel_launch(void* const* d_in, const int* in_sizes, int n_in,
                              void* d_out, int out_size, void* d_ws, size_t ws_size,
                              hipStream_t stream) {
    const float* x    = (const float*)d_in[0];   // [32,64,64,64]
    const float* wgt  = (const float*)d_in[1];   // [128,64,3,3]
    const float* bias = (const float*)d_in[2];   // [128]

    float* y   = (float*)d_out;                  // [32,128,62,62]
    float* dwo = (float*)d_out + Y_ELEMS;        // [128,64,3,3]

    char* ws = (char*)d_ws;
    __bf16* xbf   = (__bf16*)(ws + WS_XBF);
    __bf16* bpack = (__bf16*)(ws + WS_BPACK);
    float*  norms = (float*)(ws + WS_NORMS);
    float*  rsum  = (float*)(ws + WS_RSUM);
    float*  r2sum = (float*)(ws + WS_R2SUM);
    __bf16* crbf  = (__bf16*)(ws + WS_CRBF);
    float*  dwacc = (float*)(ws + WS_DWACC);

    const int XN = B_SZ * C_SZ * H_SZ * W_SZ;

    hc_init_kernel<<<(DW_ELEMS + 255) / 256, 256, 0, stream>>>(dwacc, rsum, r2sum);
    hc_norm_kernel<<<O_CH, 256, 0, stream>>>(wgt, norms);
    hc_packw_kernel<<<(DW_ELEMS + 255) / 256, 256, 0, stream>>>(wgt, norms, bpack);
    hc_xconv_kernel<<<(XN + 255) / 256, 256, 0, stream>>>(x, xbf, XN);

    hc_conv_gemm_kernel<<<dim3(B_SZ, MTILES), 256, 0, stream>>>(xbf, bpack, bias, y);

    hc_softmax_stats_kernel<<<(BL + 255) / 256, 256, 0, stream>>>(y, rsum, r2sum);
    hc_cr_build_kernel<<<(BL + 255) / 256, 256, 0, stream>>>(y, rsum, crbf);

    hc_dw_gemm_kernel<<<dim3(KTOT / 16, KSPLIT), 256, 0, stream>>>(crbf, xbf, dwacc);

    hc_finalize_kernel<<<(DW_ELEMS + 255) / 256, 256, 0, stream>>>(dwacc, rsum, r2sum, wgt, dwo);
}
